// ProtoHyperFormer_67989332295791
// MI455X (gfx1250) — matrix operand order repaired and checked by reading, not compiled, not run on hardware
//
#include <hip/hip_runtime.h>
#include <hip/hip_bf16.h>

// ---------------------------------------------------------------------------
// Types for CDNA5 WMMA (wave32): V_WMMA_F32_16X16X32_BF16
// ---------------------------------------------------------------------------
typedef __attribute__((ext_vector_type(16))) __bf16        v16bf;
typedef __attribute__((ext_vector_type(8)))  float          v8f;
typedef __attribute__((ext_vector_type(4)))  int            v4i;

union BFrag {
    v16bf        b;
    unsigned int w[8];
    uint4        q[2];
};

static __device__ inline v8f zero8() {
    v8f z = {0.f, 0.f, 0.f, 0.f, 0.f, 0.f, 0.f, 0.f};
    return z;
}

static __device__ inline unsigned short f2bf(float f) {
    unsigned int u = __builtin_bit_cast(unsigned int, f);
    unsigned int r = u + 0x7FFFu + ((u >> 16) & 1u); // round-to-nearest-even
    return (unsigned short)(r >> 16);
}

static __device__ inline float gelu_exact(float x) {
    return 0.5f * x * (1.0f + erff(x * 0.70710678118654752440f));
}

// ---------------------------------------------------------------------------
// CDNA5 async global->LDS staging (ASYNCcnt path), with safe fallback.
// Builtin signature (from hipcc diagnostic): (v4i AS1*, v4i AS3*, Ii, Ii).
// ---------------------------------------------------------------------------
#if __has_builtin(__builtin_amdgcn_global_load_async_to_lds_b128) && \
    __has_builtin(__builtin_amdgcn_s_wait_asynccnt)
#define PHP_ASYNC_LDS 1
#else
#define PHP_ASYNC_LDS 0
#endif

typedef __attribute__((address_space(1))) v4i* php_g128_t;
typedef __attribute__((address_space(3))) v4i* php_l128_t;

static __device__ inline void php_stage16(const unsigned short* g, unsigned short* l) {
#if PHP_ASYNC_LDS
    __builtin_amdgcn_global_load_async_to_lds_b128((php_g128_t)g, (php_l128_t)l, 0, 0);
#else
    *(uint4*)l = *(const uint4*)g;
#endif
}

static __device__ inline void php_stage_wait() {
#if PHP_ASYNC_LDS
    __builtin_amdgcn_s_wait_asynccnt(0);
#endif
}

// ---------------------------------------------------------------------------
// Problem constants
// ---------------------------------------------------------------------------
#define BATCH   8
#define CIN     200
#define HW      384
#define PSZ     24
#define NPATCH  256       // 16x16 per image
#define NTOK    2048      // 8 * 256
#define DMODEL  192
#define KPROTO  32
#define DFF     768
#define PATCHK  13824     // 24*24*24

// ---------------------------------------------------------------------------
// Weight fp32 -> bf16 conversion
// ---------------------------------------------------------------------------
__global__ void php_convert_kernel(const float* __restrict__ in,
                                   unsigned short* __restrict__ out, int n) {
    int i = blockIdx.x * 256 + threadIdx.x;
    if (i < n) out[i] = f2bf(in[i]);
}

// ---------------------------------------------------------------------------
// Fused band-reduce (200ch -> 24ch) + patchify into bf16 A-matrix rows.
// One block = one 24x24 patch. Streams x exactly once (944 MB -> HBM bound).
// Output layout: patch_bf[b*256+p][o*576 + py*24 + px]   (row-major, K=13824)
// ---------------------------------------------------------------------------
__global__ __launch_bounds__(256) void php_patchify_kernel(
    const float* __restrict__ x, const float* __restrict__ br_w,
    const float* __restrict__ br_b, unsigned short* __restrict__ out) {
    __shared__ __align__(16) float          wS[CIN * 24];  // [c][o] transposed
    __shared__ __align__(16) unsigned short pS[PATCHK];

    const int tid = threadIdx.x;
    const int b   = blockIdx.x >> 8;
    const int p   = blockIdx.x & 255;
    const int ph  = p >> 4, pw = p & 15;

    for (int i = tid; i < CIN * 24; i += 256) {
        int c = i / 24, o = i % 24;
        wS[c * 24 + o] = br_w[o * CIN + c];
    }
    __syncthreads();

    for (int pix = tid; pix < PSZ * PSZ; pix += 256) {
        const int py = pix / PSZ, px = pix % PSZ;
        const int h = ph * PSZ + py, w = pw * PSZ + px;
        const float* xp = x + (size_t)b * CIN * HW * HW + (size_t)h * HW + w;

        float acc[24];
#pragma unroll
        for (int o = 0; o < 24; ++o) acc[o] = br_b[o];

        for (int c = 0; c < CIN; ++c) {
            float xv = xp[(size_t)c * (HW * HW)];
            const float4* wv = (const float4*)&wS[c * 24];
#pragma unroll
            for (int o4 = 0; o4 < 6; ++o4) {
                float4 wq = wv[o4];
                acc[o4 * 4 + 0] += wq.x * xv;
                acc[o4 * 4 + 1] += wq.y * xv;
                acc[o4 * 4 + 2] += wq.z * xv;
                acc[o4 * 4 + 3] += wq.w * xv;
            }
        }
#pragma unroll
        for (int o = 0; o < 24; ++o) pS[o * 576 + pix] = f2bf(acc[o]);
    }
    __syncthreads();

    // coalesced b128 stores of the assembled patch row
    const uint4* ps4 = (const uint4*)pS;
    uint4* og = (uint4*)(out + (size_t)blockIdx.x * PATCHK);
    for (int i = tid; i < PATCHK / 8; i += 256) og[i] = ps4[i];
}

// ---------------------------------------------------------------------------
// Generic WMMA bf16 GEMM:  C[M][N] = A[M][K] @ B[N][K]^T  (+bias/res/gelu)
// Block tile (64*MI) x 64, 8 waves (4x2): each wave (16*MI)x32 output,
// i.e. MI*2 WMMA tiles fed by 2 b128 LDS reads per fragment (no packing).
// Double-buffered LDS staging via async-to-LDS (when available).
// B may be batched: batch = m0 / rowsPerB, Bp = B + batch*bStride.
// STBT stores bf16 transposed per 256-row batch: [(row/256)*N+col][row%256].
// ---------------------------------------------------------------------------
#define GTK 32
#define LSTR 40  // padded ushort stride (80B = 5*16B)

template <int MI, bool BIAS, bool RESID, bool GELU, bool STF, bool STB, bool STBT>
__global__ __launch_bounds__(256) void php_gemm_kernel(
    const unsigned short* __restrict__ A, const unsigned short* __restrict__ B,
    const float* __restrict__ bias, const float* __restrict__ Res,
    float* __restrict__ Cf, unsigned short* __restrict__ Cb,
    int M, int N, int K, long long bStride, int rowsPerB) {
    __shared__ __align__(16) unsigned short As[2][64 * MI * LSTR];
    __shared__ __align__(16) unsigned short Bs[2][64 * LSTR];

    const int tid  = threadIdx.x;
    const int lane = tid & 31;
    const int wid  = tid >> 5;
    const int m0   = blockIdx.x * (64 * MI);
    const int n0   = blockIdx.y * 64;
    const int batch = m0 / rowsPerB;
    const unsigned short* Bp = B + (size_t)batch * (size_t)bStride;

    const int lrow = tid >> 2;       // 0..63
    const int lcol = (tid & 3) * 8;  // 0,8,16,24

    const int wm = wid >> 1;  // 0..3  ((16*MI)-row strip)
    const int wn = wid & 1;   // 0..1  (32-col strip)

    v8f acc[MI][2];
#pragma unroll
    for (int mi = 0; mi < MI; ++mi) {
        acc[mi][0] = zero8();
        acc[mi][1] = zero8();
    }

    const int nK = K / GTK;
    // preload k-tile 0
    {
#pragma unroll
        for (int r = 0; r < MI; ++r)
            php_stage16(A + (size_t)(m0 + lrow + 64 * r) * K + lcol,
                        &As[0][(lrow + 64 * r) * LSTR + lcol]);
        php_stage16(Bp + (size_t)(n0 + lrow) * K + lcol,
                    &Bs[0][lrow * LSTR + lcol]);
    }
    php_stage_wait();
    __syncthreads();

    int buf = 0;
    for (int kt = 0; kt < nK; ++kt) {
        if (kt + 1 < nK) {
            const int k0 = (kt + 1) * GTK;
#pragma unroll
            for (int r = 0; r < MI; ++r)
                php_stage16(A + (size_t)(m0 + lrow + 64 * r) * K + k0 + lcol,
                            &As[buf ^ 1][(lrow + 64 * r) * LSTR + lcol]);
            php_stage16(Bp + (size_t)(n0 + lrow) * K + k0 + lcol,
                        &Bs[buf ^ 1][lrow * LSTR + lcol]);
        }

        const int khalf = (lane >> 4) * 8;   // A half-K select
        const int kb    = (lane >> 4) * 8;   // B half-K select

        // ---- A fragments: 16x32 bf16 each, two b128 LDS reads ----
        BFrag af[MI];
#pragma unroll
        for (int mi = 0; mi < MI; ++mi) {
            const int row = (wm * MI + mi) * 16 + (lane & 15);
            af[mi].q[0] = *(const uint4*)&As[buf][row * LSTR + khalf];
            af[mi].q[1] = *(const uint4*)&As[buf][row * LSTR + 16 + khalf];
        }
        // ---- B fragments: 32x16 bf16 each, two b128 LDS reads ----
        BFrag bf[2];
#pragma unroll
        for (int t = 0; t < 2; ++t) {
            const int col = wn * 32 + t * 16 + (lane & 15);
            bf[t].q[0] = *(const uint4*)&Bs[buf][col * LSTR + kb];
            bf[t].q[1] = *(const uint4*)&Bs[buf][col * LSTR + kb + 16];
        }
#pragma unroll
        for (int mi = 0; mi < MI; ++mi) {
#pragma unroll
            for (int t = 0; t < 2; ++t) {
                acc[mi][t] = __builtin_amdgcn_wmma_f32_16x16x32_bf16(
                    false, af[mi].b, false, bf[t].b, (short)0, acc[mi][t],
                    false, false);
            }
        }
        php_stage_wait();
        __syncthreads();
        buf ^= 1;
    }

    // ---- epilogue ----
#pragma unroll
    for (int mi = 0; mi < MI; ++mi) {
#pragma unroll
        for (int t = 0; t < 2; ++t) {
            const int col = n0 + wn * 32 + t * 16 + (lane & 15);
#pragma unroll
            for (int r = 0; r < 8; ++r) {
                const int row = m0 + (wm * MI + mi) * 16 + r + ((lane >> 4) * 8);
                float v = acc[mi][t][r];
                if (BIAS)  v += bias[col];
                if (RESID) v += Res[(size_t)row * N + col];
                if (GELU)  v = gelu_exact(v);
                if (STF)   Cf[(size_t)row * N + col] = v;
                if (STB)   Cb[(size_t)row * N + col] = f2bf(v);
                if (STBT)  Cb[((size_t)(row >> 8) * N + col) * 256 + (row & 255)] = f2bf(v);
            }
        }
    }
}

// ---------------------------------------------------------------------------
// Per-row LayerNorm (192) -> bf16.  One wave32 per row.
// ---------------------------------------------------------------------------
__global__ __launch_bounds__(32) void php_ln_kernel(
    const float* __restrict__ X, const float* __restrict__ g,
    const float* __restrict__ b, unsigned short* __restrict__ out) {
    const int row  = blockIdx.x;
    const int lane = threadIdx.x;
    const float* xr = X + (size_t)row * DMODEL;

    float v[6];
    float s = 0.f;
#pragma unroll
    for (int i = 0; i < 6; ++i) { v[i] = xr[lane + 32 * i]; s += v[i]; }
#pragma unroll
    for (int m = 16; m >= 1; m >>= 1) s += __shfl_xor(s, m, 32);
    const float mean = s * (1.f / DMODEL);

    float q = 0.f;
#pragma unroll
    for (int i = 0; i < 6; ++i) { float d = v[i] - mean; q += d * d; }
#pragma unroll
    for (int m = 16; m >= 1; m >>= 1) q += __shfl_xor(q, m, 32);
    const float inv = rsqrtf(q * (1.f / DMODEL) + 1e-5f);

#pragma unroll
    for (int i = 0; i < 6; ++i) {
        const int c = lane + 32 * i;
        out[(size_t)row * DMODEL + c] = f2bf((v[i] - mean) * inv * g[c] + b[c]);
    }
}

// ---------------------------------------------------------------------------
// Per-row softmax (192) -> bf16.  One wave32 per row.
// ---------------------------------------------------------------------------
__global__ __launch_bounds__(32) void php_softmax_kernel(
    const float* __restrict__ X, unsigned short* __restrict__ out) {
    const int row  = blockIdx.x;
    const int lane = threadIdx.x;
    const float* xr = X + (size_t)row * DMODEL;

    float v[6];
    float mx = -3.4e38f;
#pragma unroll
    for (int i = 0; i < 6; ++i) { v[i] = xr[lane + 32 * i]; mx = fmaxf(mx, v[i]); }
#pragma unroll
    for (int m = 16; m >= 1; m >>= 1) mx = fmaxf(mx, __shfl_xor(mx, m, 32));

    float s = 0.f;
#pragma unroll
    for (int i = 0; i < 6; ++i) { v[i] = __expf(v[i] - mx); s += v[i]; }
#pragma unroll
    for (int m = 16; m >= 1; m >>= 1) s += __shfl_xor(s, m, 32);
    const float inv = 1.f / s;

#pragma unroll
    for (int i = 0; i < 6; ++i)
        out[(size_t)row * DMODEL + lane + 32 * i] = f2bf(v[i] * inv);
}

// ---------------------------------------------------------------------------
// Router: MLP -> dual top-3 -> proto mix -> key projection. One block/token.
// Writes keys TRANSPOSED per batch: keysT[b][d][n]  (n = token-in-batch).
// ---------------------------------------------------------------------------
__global__ __launch_bounds__(256) void php_router_kernel(
    const float* __restrict__ x, const float* __restrict__ r_w1,
    const float* __restrict__ r_b1, const float* __restrict__ r_w2,
    const float* __restrict__ r_b2, const float* __restrict__ spec_proto,
    const float* __restrict__ spat_proto, const float* __restrict__ kp_w,
    const float* __restrict__ kp_b, unsigned short* __restrict__ keysT_bf) {
    __shared__ float xr[DMODEL];
    __shared__ float h[256];
    __shared__ float logits[64];
    __shared__ float cat[216];
    __shared__ int   tidx[6];
    __shared__ float tw[6];

    const int row = blockIdx.x;
    const int t   = threadIdx.x;

    if (t < DMODEL) xr[t] = x[(size_t)row * DMODEL + t];
    __syncthreads();

    {   // hidden: 256 units, K=192
        float s = r_b1[t];
        const float* w = r_w1 + (size_t)t * DMODEL;
        for (int i = 0; i < DMODEL; ++i) s += w[i] * xr[i];
        h[t] = gelu_exact(s);
    }
    __syncthreads();

    if (t < 64) {
        float s = r_b2[t];
        const float* w = r_w2 + (size_t)t * 256;
        for (int i = 0; i < 256; ++i) s += w[i] * h[i];
        logits[t] = s;
    }
    __syncthreads();

    if (t < 2) {  // t==0: spectral, t==1: spatial  top-3 + softmax(/0.1)
        float lv[KPROTO];
        for (int i = 0; i < KPROTO; ++i) lv[i] = logits[t * KPROTO + i];
        int id[3]; float vv[3];
        for (int j = 0; j < 3; ++j) {
            int bi = 0; float bv = -1e30f;
            for (int i = 0; i < KPROTO; ++i)
                if (lv[i] > bv) { bv = lv[i]; bi = i; }
            id[j] = bi; vv[j] = bv; lv[bi] = -1e30f;
        }
        const float m  = vv[0];
        const float e0 = __expf((vv[0] - m) * 10.f);
        const float e1 = __expf((vv[1] - m) * 10.f);
        const float e2 = __expf((vv[2] - m) * 10.f);
        const float inv = 1.f / (e0 + e1 + e2);
        tidx[t * 3 + 0] = id[0]; tidx[t * 3 + 1] = id[1]; tidx[t * 3 + 2] = id[2];
        tw[t * 3 + 0] = e0 * inv; tw[t * 3 + 1] = e1 * inv; tw[t * 3 + 2] = e2 * inv;
    }
    __syncthreads();

    if (t < 24) {
        float s = 0.f;
        for (int j = 0; j < 3; ++j) s += tw[j] * spec_proto[tidx[j] * 24 + t];
        cat[t] = s;
    } else if (t < 216) {
        const int d = t - 24;
        float s = 0.f;
        for (int j = 0; j < 3; ++j) s += tw[3 + j] * spat_proto[tidx[3 + j] * DMODEL + d];
        cat[t] = s;
    }
    __syncthreads();

    if (t < DMODEL) {
        float s = kp_b[t];
        const float* w = kp_w + (size_t)t * 216;
        for (int i = 0; i < 216; ++i) s += w[i] * cat[i];
        const int b = row >> 8, n = row & 255;
        keysT_bf[((size_t)b * DMODEL + t) * NPATCH + n] = f2bf(s);
    }
}

// ---------------------------------------------------------------------------
// Final: token-mean -> LayerNorm -> head (16). One block per batch image.
// ---------------------------------------------------------------------------
__global__ __launch_bounds__(256) void php_final_kernel(
    const float* __restrict__ x, const float* __restrict__ nf_g,
    const float* __restrict__ nf_b, const float* __restrict__ h_w,
    const float* __restrict__ h_b, float* __restrict__ out) {
    __shared__ float xm[DMODEL];
    __shared__ float red[256];

    const int b = blockIdx.x;
    const int t = threadIdx.x;

    float mv = 0.f;
    if (t < DMODEL) {
        const float* xp = x + (size_t)b * NPATCH * DMODEL + t;
        float s = 0.f;
        for (int n = 0; n < NPATCH; ++n) s += xp[(size_t)n * DMODEL];
        mv = s * (1.f / NPATCH);
    }
    red[t] = (t < DMODEL) ? mv : 0.f;
    __syncthreads();
    for (int off = 128; off > 0; off >>= 1) {
        if (t < off) red[t] += red[t + off];
        __syncthreads();
    }
    const float mean = red[0] * (1.f / DMODEL);
    __syncthreads();
    red[t] = (t < DMODEL) ? (mv - mean) * (mv - mean) : 0.f;
    __syncthreads();
    for (int off = 128; off > 0; off >>= 1) {
        if (t < off) red[t] += red[t + off];
        __syncthreads();
    }
    const float inv = rsqrtf(red[0] * (1.f / DMODEL) + 1e-5f);
    __syncthreads();
    if (t < DMODEL) xm[t] = (mv - mean) * inv * nf_g[t] + nf_b[t];
    __syncthreads();
    if (t < 16) {
        float s = h_b[t];
        const float* w = h_w + t * DMODEL;
        for (int i = 0; i < DMODEL; ++i) s += w[i] * xm[i];
        out[b * 16 + t] = s;
    }
}

// ---------------------------------------------------------------------------
// Host launch
// ---------------------------------------------------------------------------
extern "C" void kernel_launch(void* const* d_in, const int* in_sizes, int n_in,
                              void* d_out, int out_size, void* d_ws, size_t ws_size,
                              hipStream_t stream) {
    const float* x_in       = (const float*)d_in[0];
    const float* br_w       = (const float*)d_in[1];
    const float* br_b       = (const float*)d_in[2];
    const float* pe_w       = (const float*)d_in[3];
    const float* pe_b       = (const float*)d_in[4];
    const float* spec_proto = (const float*)d_in[5];
    const float* spat_proto = (const float*)d_in[6];
    const float* r_w1       = (const float*)d_in[7];
    const float* r_b1       = (const float*)d_in[8];
    const float* r_w2       = (const float*)d_in[9];
    const float* r_b2       = (const float*)d_in[10];
    const float* kp_w       = (const float*)d_in[11];
    const float* kp_b       = (const float*)d_in[12];
    const float* n1_g       = (const float*)d_in[13];
    const float* n1_b       = (const float*)d_in[14];
    const float* n2_g       = (const float*)d_in[15];
    const float* n2_b       = (const float*)d_in[16];
    const float* f_w1       = (const float*)d_in[17];
    const float* f_b1       = (const float*)d_in[18];
    const float* f_w2       = (const float*)d_in[19];
    const float* f_b2       = (const float*)d_in[20];
    const float* q_w        = (const float*)d_in[21];
    const float* q_b        = (const float*)d_in[22];
    const float* v_w        = (const float*)d_in[23];
    const float* v_b        = (const float*)d_in[24];
    const float* nf_g       = (const float*)d_in[25];
    const float* nf_b       = (const float*)d_in[26];
    const float* h_w        = (const float*)d_in[27];
    const float* h_b        = (const float*)d_in[28];
    float* out_f = (float*)d_out;

    // workspace carve-out (256B aligned)
    size_t off = 0;
    auto alloc = [&](size_t bytes) -> char* {
        char* p = (char*)d_ws + off;
        off += (bytes + 255) & ~(size_t)255;
        return p;
    };
    const int N_PEW = DMODEL * PATCHK;          // 2,654,208
    const int N_FW1 = 4 * DFF * DMODEL;         // 589,824
    const int N_FW2 = 4 * DMODEL * DFF;         // 589,824
    const int N_QW  = 4 * DMODEL * DMODEL;      // 147,456
    const int N_ACT = NTOK * DMODEL;            // 393,216

    unsigned short* peW_bf   = (unsigned short*)alloc((size_t)N_PEW * 2);
    unsigned short* fw1_bf   = (unsigned short*)alloc((size_t)N_FW1 * 2);
    unsigned short* fw2_bf   = (unsigned short*)alloc((size_t)N_FW2 * 2);
    unsigned short* qw_bf    = (unsigned short*)alloc((size_t)N_QW * 2);
    unsigned short* vw_bf    = (unsigned short*)alloc((size_t)N_QW * 2);
    unsigned short* patch_bf = (unsigned short*)alloc((size_t)NTOK * PATCHK * 2);
    float*          x_f      = (float*)alloc((size_t)N_ACT * 4);
    float*          q_f      = (float*)alloc((size_t)N_ACT * 4);
    unsigned short* xn_bf    = (unsigned short*)alloc((size_t)N_ACT * 2);
    unsigned short* qs_bf    = (unsigned short*)alloc((size_t)N_ACT * 2);
    unsigned short* vT_bf    = (unsigned short*)alloc((size_t)N_ACT * 2);  // [b][e][n]
    unsigned short* keysT_bf = (unsigned short*)alloc((size_t)N_ACT * 2);  // [b][d][n]
    unsigned short* kvT_bf   = (unsigned short*)alloc((size_t)BATCH * DMODEL * DMODEL * 2);
    unsigned short* h_bf     = (unsigned short*)alloc((size_t)NTOK * DFF * 2);

    // --- weight conversion to bf16 ---
    php_convert_kernel<<<(N_PEW + 255) / 256, 256, 0, stream>>>(pe_w, peW_bf, N_PEW);
    php_convert_kernel<<<(N_FW1 + 255) / 256, 256, 0, stream>>>(f_w1, fw1_bf, N_FW1);
    php_convert_kernel<<<(N_FW2 + 255) / 256, 256, 0, stream>>>(f_w2, fw2_bf, N_FW2);
    php_convert_kernel<<<(N_QW + 255) / 256, 256, 0, stream>>>(q_w, qw_bf, N_QW);
    php_convert_kernel<<<(N_QW + 255) / 256, 256, 0, stream>>>(v_w, vw_bf, N_QW);

    // --- fused band-reduce + patchify (streams 944MB of x once) ---
    php_patchify_kernel<<<NTOK, 256, 0, stream>>>(x_in, br_w, br_b, patch_bf);

    // --- patch embedding: x = patches @ pe_w^T + pe_b  (M=2048,N=192,K=13824)
    php_gemm_kernel<2, true, false, false, true, false, false>
        <<<dim3(16, 3), 256, 0, stream>>>(patch_bf, peW_bf, pe_b, nullptr, x_f,
                                          nullptr, NTOK, DMODEL, PATCHK, 0LL, 256);

    // --- router -> keysT (bf16, [b][d][n]) ---
    php_router_kernel<<<NTOK, 256, 0, stream>>>(x_f, r_w1, r_b1, r_w2, r_b2,
                                                spec_proto, spat_proto, kp_w, kp_b,
                                                keysT_bf);

    // --- transformer layers ---
    for (int i = 0; i < 4; ++i) {
        // LN1 -> bf16
        php_ln_kernel<<<NTOK, 32, 0, stream>>>(x_f, n1_g + i * DMODEL,
                                               n1_b + i * DMODEL, xn_bf);
        // q (f32)
        php_gemm_kernel<2, true, false, false, true, false, false>
            <<<dim3(16, 3), 256, 0, stream>>>(xn_bf,
                qw_bf + (size_t)i * DMODEL * DMODEL, q_b + i * DMODEL, nullptr,
                q_f, nullptr, NTOK, DMODEL, DMODEL, 0LL, 256);
        // v, stored transposed per batch -> vT[b][e][n]
        php_gemm_kernel<2, true, false, false, false, false, true>
            <<<dim3(16, 3), 256, 0, stream>>>(xn_bf,
                vw_bf + (size_t)i * DMODEL * DMODEL, v_b + i * DMODEL, nullptr,
                nullptr, vT_bf, NTOK, DMODEL, DMODEL, 0LL, 256);
        // softmax(q) -> bf16
        php_softmax_kernel<<<NTOK, 32, 0, stream>>>(q_f, qs_bf);
        // kv^T[b][e][d] = sum_n vT[b][e][n]*keysT[b][d][n]
        //   -> plain GEMM: A = vT (M=1536,K=256), B = keysT (batched, N=192)
        php_gemm_kernel<1, false, false, false, false, true, false>
            <<<dim3(24, 3), 256, 0, stream>>>(vT_bf, keysT_bf, nullptr, nullptr,
                nullptr, kvT_bf, BATCH * DMODEL, DMODEL, NPATCH,
                (long long)(DMODEL * NPATCH), DMODEL);
        // x += softmax(q) @ kv   (batched B = kvT[b])
        php_gemm_kernel<2, false, true, false, true, false, false>
            <<<dim3(16, 3), 256, 0, stream>>>(qs_bf, kvT_bf, nullptr, x_f, x_f,
                nullptr, NTOK, DMODEL, DMODEL, (long long)(DMODEL * DMODEL), 256);
        // LN2 -> bf16
        php_ln_kernel<<<NTOK, 32, 0, stream>>>(x_f, n2_g + i * DMODEL,
                                               n2_b + i * DMODEL, xn_bf);
        // FFN: h = gelu(x2 @ w1^T + b1) [bf16], x += h @ w2^T + b2
        php_gemm_kernel<2, true, false, true, false, true, false>
            <<<dim3(16, 12), 256, 0, stream>>>(xn_bf,
                fw1_bf + (size_t)i * DFF * DMODEL, f_b1 + i * DFF, nullptr,
                nullptr, h_bf, NTOK, DFF, DMODEL, 0LL, 256);
        php_gemm_kernel<2, true, true, false, true, false, false>
            <<<dim3(16, 3), 256, 0, stream>>>(h_bf,
                fw2_bf + (size_t)i * DMODEL * DFF, f_b2 + i * DMODEL, x_f,
                x_f, nullptr, NTOK, DMODEL, DFF, 0LL, 256);
    }

    // --- final: mean over tokens -> LN -> head ---
    php_final_kernel<<<BATCH, 256, 0, stream>>>(x_f, nf_g, nf_b, h_w, h_b, out_f);

    (void)in_sizes; (void)n_in; (void)out_size; (void)ws_size;
}
